// Model_46273977647650
// MI455X (gfx1250) — matrix operand order repaired and checked by reading, not compiled, not run on hardware
//
#include <hip/hip_runtime.h>
#include <hip/hip_bf16.h>

// Problem constants (reference: B=64, T=256, F=16, H=128, K=3)
#define BN 64
#define TN 256
#define FN 16
#define HN 128
#define G4 512   // 4*H
#define MWG 16   // batches per workgroup (WMMA M tile)

typedef __attribute__((ext_vector_type(16))) __bf16 v16bf;
typedef __attribute__((ext_vector_type(8)))  __bf16 v8bf;
typedef __attribute__((ext_vector_type(4)))  __bf16 v4bf;
typedef __attribute__((ext_vector_type(8)))  float  v8f;

__device__ __forceinline__ float fast_tanh(float x) {
#if __has_builtin(__builtin_amdgcn_tanhf)
    return __builtin_amdgcn_tanhf(x);          // v_tanh_f32 on gfx1250
#else
    float e = __expf(2.0f * x);
    return (e - 1.0f) * __builtin_amdgcn_rcpf(e + 1.0f);
#endif
}
__device__ __forceinline__ float fast_sigmoid(float x) {
    return fmaf(0.5f, fast_tanh(0.5f * x), 0.5f);
}

// ---------------------------------------------------------------------------
// Kernel 1: per-(feature, batch-group-of-16) LSTM over the ragged time slice.
// grid = (4 batch groups, 16 features), block = 256 threads (8 wave32).
// Each wave owns one 16-column h-slice across ALL FOUR gate groups, so the
// entire cell update (sigmoid/tanh, c, h) lives in registers; h ping-pongs
// through LDS (bf16) for the next step's A fragments. One barrier per step.
// ---------------------------------------------------------------------------
__launch_bounds__(256, 1)
__global__ void lstm_feature_kernel(const float* __restrict__ x,      // [B,T,F]
                                    const int*   __restrict__ index,  // [B,2]
                                    const float* __restrict__ W_ih,   // [F,4H]
                                    const float* __restrict__ W_hh,   // [F,4H,H]
                                    const float* __restrict__ bias,   // [F,4H]
                                    float*       __restrict__ hfinal) // ws [B,F,H]
{
    __shared__ __align__(32) __bf16 w_lds[G4 * HN];       // 128 KB W_hh[f] bf16
    __shared__ __align__(32) __bf16 h_lds[2][MWG * HN];   // 8 KB ping-pong h
    __shared__ __align__(16) float  x_all[TN * MWG];      // 16 KB x slab
    __shared__ int s_start[MWG], s_end[MWG];
    __shared__ int s_tlo, s_thi;

    const int mg  = blockIdx.x;        // batch group 0..3
    const int f   = blockIdx.y;        // feature 0..15
    const int b0  = mg * MWG;
    const int tid = threadIdx.x;

    // ---- preload W_hh[f] (f32 -> bf16) into LDS, vectorized float4 ----
    const float4* wsrc = (const float4*)(W_hh + (size_t)f * G4 * HN);
    for (int i = tid; i < G4 * HN / 4; i += 256) {
        float4 v = wsrc[i];
        v4bf p;
        p[0] = (__bf16)v.x; p[1] = (__bf16)v.y;
        p[2] = (__bf16)v.z; p[3] = (__bf16)v.w;
        *(v4bf*)(&w_lds[i * 4]) = p;
    }
    // ---- preload all x for this (feature, batch group): x_all[t*16+m] ----
    for (int i = tid; i < TN * MWG; i += 256) {
        int tt = i >> 4, m = i & 15;
        x_all[i] = x[(size_t)(b0 + m) * TN * FN + tt * FN + f];
    }
    for (int i = tid; i < 2 * MWG * HN; i += 256)
        ((__bf16*)h_lds)[i] = (__bf16)0.0f;
    if (tid < MWG) {
        s_start[tid] = index[2 * (b0 + tid) + 0];
        s_end[tid]   = index[2 * (b0 + tid) + 1];
    }
    // zero this WG's slice of hfinal (ws is poisoned by the harness)
    for (int i = tid; i < MWG * HN; i += 256)
        hfinal[(size_t)(b0 + (i >> 7)) * (FN * HN) + f * HN + (i & (HN - 1))] = 0.0f;
    __syncthreads();
    if (tid == 0) {
        int lo = TN, hi = 0;
        for (int bb = 0; bb < MWG; ++bb) {
            lo = (s_start[bb] < lo) ? s_start[bb] : lo;
            hi = (s_end[bb]   > hi) ? s_end[bb]   : hi;
        }
        s_tlo = lo; s_thi = hi;
    }
    __syncthreads();
    const int t_lo = s_tlo, t_hi = s_thi;

    const int wave  = tid >> 5;
    const int lane  = tid & 31;
    const int lm    = lane & 15;
    const int hcol  = wave * 16 + lm;        // this wave's h-slice column
    const int koffA = (lane < 16) ? 0 : 16;  // A frag K: 0-15 / 16-31 contiguous
    const int koffB = (lane < 16) ? 0 : 16;  // B frag K: 0-15 / 16-31 contiguous
    const int mhi   = (lane >> 4) << 3;      // C/D: VGPR r -> M = r + mhi

    // per-lane time-invariant params (W_ih, bias per gate; slice bounds per m)
    float wihv[4], bv[4];
    #pragma unroll
    for (int g = 0; g < 4; ++g) {
        wihv[g] = W_ih[f * G4 + g * HN + hcol];
        bv[g]   = bias[f * G4 + g * HN + hcol];
    }
    int stv[8], env[8];
    #pragma unroll
    for (int r = 0; r < 8; ++r) { stv[r] = s_start[mhi + r]; env[r] = s_end[mhi + r]; }

    float c[8], hreg[8];
    #pragma unroll
    for (int r = 0; r < 8; ++r) { c[r] = 0.0f; hreg[r] = 0.0f; }

    int p = 0;
    for (int t = t_lo; t < t_hi; ++t) {
        const __bf16* hr = h_lds[p];
        __bf16*       hw = h_lds[p ^ 1];

        // x for this step (8 consecutive floats per lane half)
        float4 xq0 = *(const float4*)&x_all[t * MWG + mhi + 0];
        float4 xq1 = *(const float4*)&x_all[t * MWG + mhi + 4];
        float xv[8] = { xq0.x, xq0.y, xq0.z, xq0.w, xq1.x, xq1.y, xq1.z, xq1.w };

        // A fragments from previous h
        v16bf A[4];
        #pragma unroll
        for (int k = 0; k < 4; ++k) {
            const __bf16* ap = &hr[lm * HN + k * 32 + koffA];
            v8bf lo = *(const v8bf*)ap;
            v8bf hi = *(const v8bf*)(ap + 8);
            A[k] = __builtin_shufflevector(lo, hi,
                    0, 1, 2, 3, 4, 5, 6, 7, 8, 9, 10, 11, 12, 13, 14, 15);
        }

        // seed accumulators with x*W_ih + bias (i,f,g,o at this h-slice)
        v8f acc[4];
        #pragma unroll
        for (int g = 0; g < 4; ++g) {
            #pragma unroll
            for (int r = 0; r < 8; ++r) acc[g][r] = fmaf(xv[r], wihv[g], bv[g]);
        }

        // 16 WMMAs, k outer / gate inner -> RAW distance 4 on each accumulator
        #pragma unroll
        for (int k = 0; k < 4; ++k) {
            #pragma unroll
            for (int g = 0; g < 4; ++g) {
                const __bf16* bp = &w_lds[(g * HN + hcol) * HN + k * 32 + koffB];
                v16bf Bf = *(const v16bf*)bp;
                acc[g] = __builtin_amdgcn_wmma_f32_16x16x32_bf16(
                            false, A[k], false, Bf, (short)0, acc[g], false, false);
            }
        }

        // LSTM cell update fully in registers
        #pragma unroll
        for (int r = 0; r < 8; ++r) {
            if (t >= stv[r] && t < env[r]) {
                float ig = fast_sigmoid(acc[0][r]);
                float fg = fast_sigmoid(acc[1][r]);
                float gg = fast_tanh   (acc[2][r]);
                float og = fast_sigmoid(acc[3][r]);
                float cn = fmaf(fg, c[r], ig * gg);
                float hn = og * fast_tanh(cn);
                c[r]    = cn;
                hreg[r] = hn;
                if (t == env[r] - 1)   // last valid step: emit f32 h
                    hfinal[(size_t)(b0 + mhi + r) * (FN * HN) + f * HN + hcol] = hn;
            }
            hw[(mhi + r) * HN + hcol] = (__bf16)hreg[r];
        }

        p ^= 1;
        __syncthreads();   // h ready for next step's A fragments
    }
}

// ---------------------------------------------------------------------------
// Kernel 2: per-batch head. mean over H -> conv1d(K=3,pad=1) over F ->
// sigmoid gate -> scale h -> FC [F*H,2]. grid = B, block = 256.
// ---------------------------------------------------------------------------
__launch_bounds__(256, 1)
__global__ void head_kernel(const float* __restrict__ hfinal, // [B,F,H]
                            const float* __restrict__ conv_w, // [3]
                            const float* __restrict__ fc_w,   // [F*H,2]
                            const float* __restrict__ fc_b,   // [2]
                            float*       __restrict__ out)    // [B,2]
{
    __shared__ float hbuf[FN * HN];
    __shared__ float fmean[FN];
    __shared__ float gatef[FN];
    __shared__ float red0[256];
    __shared__ float red1[256];

    const int b   = blockIdx.x;
    const int tid = threadIdx.x;

    for (int i = tid; i < FN * HN; i += 256)
        hbuf[i] = hfinal[(size_t)b * (FN * HN) + i];
    __syncthreads();

    if (tid < FN) {
        float s = 0.0f;
        for (int h = 0; h < HN; ++h) s += hbuf[tid * HN + h];
        fmean[tid] = s * (1.0f / (float)HN);
    }
    __syncthreads();
    if (tid < FN) {
        float a = 0.0f;
        #pragma unroll
        for (int k = 0; k < 3; ++k) {
            int ff = tid + k - 1;
            if (ff >= 0 && ff < FN) a += fmean[ff] * conv_w[k];
        }
        gatef[tid] = fast_sigmoid(a);
    }
    __syncthreads();

    float a0 = 0.0f, a1 = 0.0f;
    for (int i = tid; i < FN * HN; i += 256) {
        float v = hbuf[i] * gatef[i >> 7];
        a0 = fmaf(v, fc_w[2 * i + 0], a0);
        a1 = fmaf(v, fc_w[2 * i + 1], a1);
    }
    red0[tid] = a0; red1[tid] = a1;
    __syncthreads();
    for (int s = 128; s > 0; s >>= 1) {
        if (tid < s) { red0[tid] += red0[tid + s]; red1[tid] += red1[tid + s]; }
        __syncthreads();
    }
    if (tid == 0) {
        out[2 * b + 0] = red0[0] + fc_b[0];
        out[2 * b + 1] = red1[0] + fc_b[1];
    }
}

// ---------------------------------------------------------------------------
extern "C" void kernel_launch(void* const* d_in, const int* in_sizes, int n_in,
                              void* d_out, int out_size, void* d_ws, size_t ws_size,
                              hipStream_t stream)
{
    const float* x      = (const float*)d_in[0];  // [64,256,16]
    const int*   index  = (const int*)  d_in[1];  // [64,2]
    const float* W_ih   = (const float*)d_in[2];  // [16,512]
    const float* W_hh   = (const float*)d_in[3];  // [16,512,128]
    const float* bias   = (const float*)d_in[4];  // [16,512]
    const float* conv_w = (const float*)d_in[5];  // [3]
    const float* fc_w   = (const float*)d_in[6];  // [2048,2]
    const float* fc_b   = (const float*)d_in[7];  // [2]
    float*       out    = (float*)d_out;          // [64,2]
    float*       hfin   = (float*)d_ws;           // needs 64*16*128*4 = 512 KB

    dim3 grid1(BN / MWG, FN);   // 4 batch groups x 16 features = 64 WGs
    lstm_feature_kernel<<<grid1, 256, 0, stream>>>(x, index, W_ih, W_hh, bias, hfin);
    head_kernel<<<BN, 256, 0, stream>>>(hfin, conv_w, fc_w, fc_b, out);
}
